// PVRNN_55731495633105
// MI455X (gfx1250) — matrix operand order repaired and checked by reading, not compiled, not run on hardware
//
#include <hip/hip_runtime.h>
#include <hip/hip_bf16.h>

// ---------------------------------------------------------------------------
// PVRNN persistent-scan kernel for gfx1250 (MI455X).
//   B=128, T=2048, D0=D1=512, Z0=64, Z1=32, OUT=64, TAU0=2, TAU1=8
// 38 persistent workgroups, f16 WMMA (v_wmma_f32_16x16x32_f16), weights
// pre-transposed to f16 in LDS, d-state ping-pong in L2-resident scratch,
// grid-wide sense-reversing barrier with agent-scope fences.
// All global traffic through addrspace(1) pointers -> global_load/store_*,
// all transcendentals branchless (v_tanh_f32 / v_exp_f32 / v_rcp_f32).
// ---------------------------------------------------------------------------

typedef __attribute__((ext_vector_type(16))) _Float16 v16h;
typedef __attribute__((ext_vector_type(8)))  _Float16 v8h;
typedef __attribute__((ext_vector_type(8)))  float    v8f;
typedef __attribute__((ext_vector_type(4)))  float    v4f;

// addrspace(1) = global memory; forces global_load/store instead of flat.
typedef const __attribute__((address_space(1))) _Float16* gh_cp;
typedef       __attribute__((address_space(1))) _Float16* gh_p;
typedef const __attribute__((address_space(1))) float*    gf_cp;
typedef       __attribute__((address_space(1))) float*    gf_p;
typedef const __attribute__((address_space(1))) v8h*      gv8h_cp;
typedef const __attribute__((address_space(1))) v4f*      gv4f_cp;

#define PV_B    128
#define PV_T    2048
#define PV_D    512
#define PV_Z0   64
#define PV_Z1   32
#define PV_OUT  64
#define NBLK    38
#define NX      ((long)PV_B * PV_T * PV_OUT)   // 16777216 floats of x

// scratch layout (bytes)
#define WS_DHALVES   65536          // halves per d buffer (128*512)
#define BAR_OFF_B    524288
#define KL_OFF_B     524352
#define WS_ZERO_U32  ((524288 + 128) / 4)

// --------------------------- fast transcendentals ---------------------------

__device__ __forceinline__ float fexp(float x) {       // e^x, 1 TRANS + 1 VALU
  return __builtin_amdgcn_exp2f(x * 1.4426950408889634f);
}

__device__ __forceinline__ float ftanh(float x) {
#if __has_builtin(__builtin_amdgcn_tanhf)
  return __builtin_amdgcn_tanhf(x);                    // v_tanh_f32
#else
  float xc = fminf(9.f, fmaxf(-9.f, x));               // branchless fallback
  float e  = __builtin_amdgcn_exp2f(xc * 2.8853900817779268f);  // e^(2x)
  return 1.f - 2.f * __builtin_amdgcn_rcpf(e + 1.f);
#endif
}

__device__ __forceinline__ float fsigmoid(float x) {
  return __builtin_amdgcn_rcpf(1.f + __builtin_amdgcn_exp2f(
                                         x * -1.4426950408889634f));
}

// ------------------------------ WMMA helpers --------------------------------

__device__ __forceinline__ v8f wmma_f16(v16h a, v16h b, v8f c) {
  return __builtin_amdgcn_wmma_f32_16x16x32_f16(false, a, false, b, (short)0, c,
                                                false, false);
}

// A-fragment (16x32 f16) from row-major f16 matrix in GLOBAL memory, ld=512.
// lanes 0-15: row=lane,   K = k0+0..7  and k0+16..23
// lanes16-31: row=lane-16,K = k0+8..15 and k0+24..31
__device__ __forceinline__ v16h load_a16(const _Float16* A, int row0, int k0,
                                         int lane) {
  const _Float16* p = A + (row0 + (lane & 15)) * PV_D + k0 + ((lane >> 4) << 3);
  v16h r;
  *(v8h*)&r       = *(gv8h_cp)p;
  *((v8h*)&r + 1) = *(gv8h_cp)(p + 16);
  return r;
}

// B-fragment (32x16 f16) from LDS weight tile stored transposed [16][K].
__device__ __forceinline__ v16h load_b16(const _Float16* Wt, int K, int k0,
                                         int lane) {
  const _Float16* p = Wt + (lane & 15) * K + k0 + ((lane >> 4) << 3);
  v16h r;
  *(v8h*)&r       = *(const v8h*)p;
  *((v8h*)&r + 1) = *(const v8h*)(p + 16);
  return r;
}

// z = tanh(a_mu) + exp(tanh(a_sg)) * eps, built directly as an A-fragment.
// Per-lane reads are 32B-aligned contiguous -> global_load_b128 pairs.
__device__ __forceinline__ v16h make_z_frag(const float* AM, const float* AS,
                                            const float* EP, long tz, long TZ,
                                            int k0, int row0, int lane) {
  int r  = row0 + (lane & 15);
  int kb = k0 + ((lane >> 4) << 3);
  long o = (long)r * TZ + tz + kb;
  v16h out;
#pragma unroll
  for (int half = 0; half < 2; ++half) {
    long oo = o + half * 16;
    v4f m0 = *(gv4f_cp)(AM + oo), m1 = *(gv4f_cp)(AM + oo + 4);
    v4f s0 = *(gv4f_cp)(AS + oo), s1 = *(gv4f_cp)(AS + oo + 4);
    v4f e0 = *(gv4f_cp)(EP + oo), e1 = *(gv4f_cp)(EP + oo + 4);
#pragma unroll
    for (int i = 0; i < 4; ++i) {
      out[half * 8 + i]     = (_Float16)(ftanh(m0[i]) + fexp(ftanh(s0[i])) * e0[i]);
      out[half * 8 + 4 + i] = (_Float16)(ftanh(m1[i]) + fexp(ftanh(s1[i])) * e1[i]);
    }
  }
  return out;
}

// Stage a 16-column slice of a row-major f32 weight [K][ld] into LDS as
// f16 transposed [16][K] (so B-fragments are contiguous 16B reads).
__device__ __forceinline__ void stage16(const float* W, int ld, int c0, int K,
                                        _Float16* dst) {
  for (int idx = threadIdx.x; idx < 16 * K; idx += 256) {
    int n = idx & 15, k = idx >> 4;
    dst[n * K + k] = (_Float16)(((gf_cp)W)[(long)k * ld + c0 + n]);
  }
}

// Grid-wide sense-reversing barrier (generation counter in L2).
__device__ __forceinline__ void grid_barrier(unsigned* bar) {
  __syncthreads();
  if (threadIdx.x == 0) {
    __builtin_amdgcn_fence(__ATOMIC_RELEASE, "agent");
    unsigned g = __hip_atomic_load(&bar[1], __ATOMIC_RELAXED,
                                   __HIP_MEMORY_SCOPE_AGENT);
    unsigned a = __hip_atomic_fetch_add(&bar[0], 1u, __ATOMIC_ACQ_REL,
                                        __HIP_MEMORY_SCOPE_AGENT);
    if (a == (unsigned)(NBLK - 1)) {
      __hip_atomic_store(&bar[0], 0u, __ATOMIC_RELAXED,
                         __HIP_MEMORY_SCOPE_AGENT);
      __hip_atomic_fetch_add(&bar[1], 1u, __ATOMIC_RELEASE,
                             __HIP_MEMORY_SCOPE_AGENT);
    } else {
      while (__hip_atomic_load(&bar[1], __ATOMIC_RELAXED,
                               __HIP_MEMORY_SCOPE_AGENT) == g)
        __builtin_amdgcn_s_sleep(1);
    }
    __builtin_amdgcn_fence(__ATOMIC_ACQUIRE, "agent");
  }
  __syncthreads();
}

// ------------------------------ kernels ------------------------------------

__global__ void pvrnn_init_ws(unsigned* __restrict__ ws) {
  gf_p w = (gf_p)ws;
  for (long i = (long)blockIdx.x * blockDim.x + threadIdx.x; i < WS_ZERO_U32;
       i += (long)gridDim.x * blockDim.x)
    ((__attribute__((address_space(1))) unsigned*)ws)[i] = 0u;
  (void)w;
}

__global__ void __launch_bounds__(256)
pvrnn_persistent(const float* __restrict__ Wdh0, const float* __restrict__ Wzh0,
                 const float* __restrict__ bh0, const float* __restrict__ Wmup0,
                 const float* __restrict__ bmup0, const float* __restrict__ Wsgp0,
                 const float* __restrict__ bsgp0, const float* __restrict__ Amu0,
                 const float* __restrict__ Asg0, const float* __restrict__ Eps0,
                 const float* __restrict__ Wdh1, const float* __restrict__ Wzh1,
                 const float* __restrict__ Wih1, const float* __restrict__ bh1,
                 const float* __restrict__ Wmup1, const float* __restrict__ bmup1,
                 const float* __restrict__ Wsgp1, const float* __restrict__ bsgp1,
                 const float* __restrict__ Amu1, const float* __restrict__ Asg1,
                 const float* __restrict__ Eps1, const float* __restrict__ Wout,
                 const float* __restrict__ bout, float* __restrict__ xout,
                 _Float16* __restrict__ dbuf, unsigned* __restrict__ bar,
                 float* __restrict__ klslots) {
  extern __shared__ _Float16 smem[];
  const int g     = blockIdx.x;
  const int tid   = threadIdx.x;
  const int lane  = tid & 31;
  const int wave  = tid >> 5;
  const int row0  = wave << 4;          // 8 waves cover all 128 batch rows
  const int nlane = lane & 15;
  const int mrow  = row0 + ((lane >> 4) << 3);  // row of accumulator elem v=0

  const bool roleA  = (g < 32);                 // d0n / d1n column tile g
  const bool roleP0 = (g >= 32 && g < 36);      // prior-0 z-tile g-32
  const bool roleP1 = (g >= 36);                // prior-1 z-tile g-36
  const bool roleX  = (g >= 34);                // x-tile g-34

  _Float16* d0b[2] = {dbuf, dbuf + WS_DHALVES};
  _Float16* d1b[2] = {dbuf + 2 * WS_DHALVES, dbuf + 3 * WS_DHALVES};

  // LDS carve-up (halves).
  _Float16* L0 = smem;            // Wdh0t | Wmup0t | Wmup1t   [16][512]
  _Float16* L1 = smem + 8192;     // Wzh0t(16x64) | Wsgp0t | Wsgp1t
  _Float16* L2 = smem + 9216;     // Wdh1t [16][512]
  _Float16* L3 = smem + 17408;    // Wih1t [16][512]
  _Float16* L4 = smem + 25600;    // Wzh1t [16][32]
  _Float16* LX = smem + 16384;    // Woutt [16][512]  (roleX WGs are 34..37)
  float*  redf = (float*)(smem + 26368);  // 8-float wave-reduction scratch

  if (roleA) {
    stage16(Wdh0, PV_D, 16 * g, PV_D, L0);
    stage16(Wzh0, PV_D, 16 * g, PV_Z0, L1);
    stage16(Wdh1, PV_D, 16 * g, PV_D, L2);
    stage16(Wih1, PV_D, 16 * g, PV_D, L3);
    stage16(Wzh1, PV_D, 16 * g, PV_Z1, L4);
  } else if (roleP0) {
    stage16(Wmup0, PV_Z0, 16 * (g - 32), PV_D, L0);
    stage16(Wsgp0, PV_Z0, 16 * (g - 32), PV_D, L1);
  } else {
    stage16(Wmup1, PV_Z1, 16 * (g - 36), PV_D, L0);
    stage16(Wsgp1, PV_Z1, 16 * (g - 36), PV_D, L1);
  }
  if (roleX) stage16(Wout, PV_OUT, 16 * (g - 34), PV_D, LX);
  __syncthreads();

  // per-lane biases (column-indexed)
  float biasH0 = roleA ? ((gf_cp)bh0)[16 * g + nlane] : 0.f;
  float biasH1 = roleA ? ((gf_cp)bh1)[16 * g + nlane] : 0.f;
  float biasPm = roleP0 ? ((gf_cp)bmup0)[16 * (g - 32) + nlane]
                        : (roleP1 ? ((gf_cp)bmup1)[16 * (g - 36) + nlane] : 0.f);
  float biasPs = roleP0 ? ((gf_cp)bsgp0)[16 * (g - 32) + nlane]
                        : (roleP1 ? ((gf_cp)bsgp1)[16 * (g - 36) + nlane] : 0.f);
  float biasX = roleX ? ((gf_cp)bout)[16 * (g - 34) + nlane] : 0.f;

  v8f h0 = {};           // recurrent state, lives in registers for all T
  v8f h1 = {};
  float klacc = 0.f;

  const long TZ0 = (long)PV_T * PV_Z0;
  const long TZ1 = (long)PV_T * PV_Z1;

  for (int t = 0; t < PV_T; ++t) {
    const int p = t & 1;
    const _Float16* d0r = d0b[p];
    _Float16*       d0w = d0b[1 - p];
    const _Float16* d1r = d1b[p];
    _Float16*       d1w = d1b[1 - p];

    // ---------------- phase A : layer-0 recurrence + both priors ----------
    if (roleA) {
      v8f acc = {};
#pragma unroll 4
      for (int kc = 0; kc < 16; ++kc)
        acc = wmma_f16(load_a16(d0r, row0, kc * 32, lane),
                       load_b16(L0, PV_D, kc * 32, lane), acc);
#pragma unroll
      for (int kc = 0; kc < 2; ++kc)
        acc = wmma_f16(make_z_frag(Amu0, Asg0, Eps0, (long)t * PV_Z0, TZ0,
                                   kc * 32, row0, lane),
                       load_b16(L1, PV_Z0, kc * 32, lane), acc);
#pragma unroll
      for (int v = 0; v < 8; ++v) {
        h0[v] = 0.5f * h0[v] + 0.5f * (acc[v] + biasH0);   // TAU0 = 2
        ((gh_p)d0w)[(mrow + v) * PV_D + 16 * g + nlane] =
            (_Float16)ftanh(h0[v]);
      }
    } else {
      const _Float16* ar = roleP0 ? d0r : d1r;
      v8f accM = {}, accS = {};
#pragma unroll 4
      for (int kc = 0; kc < 16; ++kc) {
        v16h a = load_a16(ar, row0, kc * 32, lane);
        accM = wmma_f16(a, load_b16(L0, PV_D, kc * 32, lane), accM);
        accS = wmma_f16(a, load_b16(L1, PV_D, kc * 32, lane), accS);
      }
      const float* AM = roleP0 ? Amu0 : Amu1;
      const float* AS = roleP0 ? Asg0 : Asg1;
      const long TZ = roleP0 ? TZ0 : TZ1;
      const long tz = (long)t * (roleP0 ? PV_Z0 : PV_Z1);
      const int zoff = roleP0 ? 16 * (g - 32) : 16 * (g - 36);
#pragma unroll
      for (int v = 0; v < 8; ++v) {
        long o    = (long)(mrow + v) * TZ + tz + zoff + nlane;
        float tp  = ftanh(accS[v] + biasPs);   // log(sgp)
        float mup = ftanh(accM[v] + biasPm);
        float tq  = ftanh(((gf_cp)AS)[o]);     // log(sgq)
        float muq = ftanh(((gf_cp)AM)[o]);
        float sgq = fexp(tq);
        float dm  = muq - mup;
        klacc += tp - tq + (sgq * sgq + dm * dm) * 0.5f * fexp(-2.f * tp)
                 - 0.5f;
      }
    }
    grid_barrier(bar);   // d0n complete

    // ---------------- phase B : layer-1 recurrence ------------------------
    if (roleA) {
      v8f acc = {};
#pragma unroll 4
      for (int kc = 0; kc < 16; ++kc)
        acc = wmma_f16(load_a16(d1r, row0, kc * 32, lane),
                       load_b16(L2, PV_D, kc * 32, lane), acc);
#pragma unroll 4
      for (int kc = 0; kc < 16; ++kc)
        acc = wmma_f16(load_a16(d0w, row0, kc * 32, lane),   // d0n (this step)
                       load_b16(L3, PV_D, kc * 32, lane), acc);
      acc = wmma_f16(make_z_frag(Amu1, Asg1, Eps1, (long)t * PV_Z1, TZ1, 0,
                                 row0, lane),
                     load_b16(L4, PV_Z1, 0, lane), acc);
#pragma unroll
      for (int v = 0; v < 8; ++v) {
        h1[v] = 0.875f * h1[v] + 0.125f * (acc[v] + biasH1);  // TAU1 = 8
        ((gh_p)d1w)[(mrow + v) * PV_D + 16 * g + nlane] =
            (_Float16)ftanh(h1[v]);
      }
    }
    grid_barrier(bar);   // d1n complete

    // ---------------- phase C : output head (overlaps next phase A) -------
    if (roleX) {
      v8f acc = {};
#pragma unroll 4
      for (int kc = 0; kc < 16; ++kc)
        acc = wmma_f16(load_a16(d1w, row0, kc * 32, lane),
                       load_b16(LX, PV_D, kc * 32, lane), acc);
      const int oc = 16 * (g - 34) + nlane;
#pragma unroll
      for (int v = 0; v < 8; ++v)
        ((gf_p)xout)[((long)(mrow + v) * PV_T + t) * PV_OUT + oc] =
            fsigmoid(acc[v] + biasX);
    }
  }

  // ---- KL reduction: wave shuffle -> LDS -> per-WG slot (deterministic) ---
#pragma unroll
  for (int off = 16; off; off >>= 1) klacc += __shfl_xor(klacc, off, 32);
  if (lane == 0) redf[wave] = klacc;
  __syncthreads();
  if (tid == 0 && g >= 32) {
    float s = 0.f;
#pragma unroll
    for (int w = 0; w < 8; ++w) s += redf[w];
    ((gf_p)klslots)[g - 32] = s / (float)PV_B;  // 0..3 = kl0 tiles, 4..5 = kl1
  }
  grid_barrier(bar);
  if (g == 0 && tid == 0) {
    gf_cp ks = (gf_cp)klslots;
    float k0 = ks[0] + ks[1] + ks[2] + ks[3];
    float k1 = ks[4] + ks[5];
    ((gf_p)xout)[NX]     = k0 + k1;                   // kl
    ((gf_p)xout)[NX + 1] = 0.001f * k0 + 0.001f * k1; // wkl (W0=W1=1e-3)
  }
}

// ------------------------------ launcher -----------------------------------

extern "C" void kernel_launch(void* const* d_in, const int* in_sizes, int n_in,
                              void* d_out, int out_size, void* d_ws,
                              size_t ws_size, hipStream_t stream) {
  (void)in_sizes; (void)n_in; (void)out_size; (void)ws_size;

  const float* in[23];
  for (int i = 0; i < 23; ++i) in[i] = (const float*)d_in[i];

  unsigned*  ws32    = (unsigned*)d_ws;
  _Float16*  dbuf    = (_Float16*)d_ws;
  unsigned*  bar     = (unsigned*)((char*)d_ws + BAR_OFF_B);
  float*     klslots = (float*)((char*)d_ws + KL_OFF_B);

  // zero d-state ping-pong buffers, barrier and kl slots (graph-safe)
  pvrnn_init_ws<<<256, 256, 0, stream>>>(ws32);

  pvrnn_persistent<<<NBLK, 256, 53248, stream>>>(
      in[0], in[1], in[2], in[3], in[4], in[5], in[6], in[7], in[8], in[9],
      in[10], in[11], in[12], in[13], in[14], in[15], in[16], in[17], in[18],
      in[19], in[20], in[21], in[22], (float*)d_out, dbuf, bar, klslots);
}
